// TripleBarrierLSTM_45913200394692
// MI455X (gfx1250) — matrix operand order repaired and checked by reading, not compiled, not run on hardware
//
#include <hip/hip_runtime.h>

// ---------------- problem constants ----------------
#define BATCH 32768
#define SEQ   20
#define NFEAT 25
#define H1    128
#define G4    512      // 4*H1
#define H2    64
#define DH    32
#define NC    3
#define BM    16       // batch rows per workgroup

// ---------------- WMMA types ----------------
typedef __attribute__((ext_vector_type(16))) __bf16        v16bf;
typedef __attribute__((ext_vector_type(8)))  float         v8f;
typedef __attribute__((ext_vector_type(4)))  unsigned int  u32x4;

union FragU { u32x4 q[2]; v16bf v; };

// ---------------- LDS layout (bytes) ----------------
#define OFF_WBUF   0            // 512*128 bf16            = 131072
#define OFF_WIH0   131072       // 512*32  bf16 (padded)   =  32768
#define OFF_H1SEQ  163840       // 20*16*128 bf16          =  81920
#define OFF_GACT   245760       // 16*512 f32              =  32768
#define OFF_C      278528       // 16*128 f32              =   8192
#define OFF_H      286720       // 16*128 bf16             =   4096
#define OFF_XBUF   290816       // 16*32  bf16             =   1024
#define OFF_B0     291840       // 512 f32                 =   2048
#define OFF_B1     293888       // 512 f32                 =   2048
#define LDS_BYTES  295936       // 289 KB  (<= 320 KB/WGP)

// ---------------- helpers ----------------
__device__ __forceinline__ unsigned short f2bf(float f) {
    union { float f; unsigned u; } v; v.f = f;
    unsigned u = v.u;
    return (unsigned short)((u + 0x7FFFu + ((u >> 16) & 1u)) >> 16);  // RNE
}
__device__ __forceinline__ float bf2f(unsigned short s) {
    union { unsigned u; float f; } v; v.u = ((unsigned)s) << 16;
    return v.f;
}
__device__ __forceinline__ unsigned pk2(float a, float b) {
    return (unsigned)f2bf(a) | ((unsigned)f2bf(b) << 16);
}
// raw v_rcp_f32 (no IEEE refine sequence); exp->inf saturates cleanly to 0
__device__ __forceinline__ float fast_rcp(float x) { return __builtin_amdgcn_rcpf(x); }
// tanh(x) = 2*sigmoid(2x) - 1, branch-free: one v_exp + one v_rcp
__device__ __forceinline__ float tanh_fast(float x) {
    return 2.0f * fast_rcp(1.0f + __expf(-2.0f * x)) - 1.0f;
}

// A fragment: 16(M) x 32(K) bf16 tile from row-major LDS [rows][stride].
// Lanes 0-15 hold M=0..15 with K = kbase+0..15; lanes 16-31 same M,
// K = kbase+16..31  (contiguous, matches B fragment order).
__device__ __forceinline__ v16bf load_afrag(const unsigned short* base, int stride,
                                            int kbase, int lane) {
    int m    = lane & 15;
    int koff = (lane >> 4) << 4;                 // 0 or 16
    FragU f;
    f.q[0] = *(const u32x4*)(base + m * stride + kbase + koff);
    f.q[1] = *(const u32x4*)(base + m * stride + kbase + koff + 8);
    return f.v;
}

// B fragment: 32(K) x 16(N) bf16 tile of W^T, i.e. rows n0..n0+15 of row-major
// W [G][stride].  Lanes 0-15 => K=k0..k0+15, lanes 16-31 => K=k0+16..k0+31;
// lane&15 selects column N.  16 contiguous bf16 per lane.
__device__ __forceinline__ v16bf load_bfrag(const unsigned short* w, int stride,
                                            int n0, int k0, int lane) {
    int n  = n0 + (lane & 15);
    int kh = k0 + ((lane >> 4) << 4);            // 0 or 16 within K block
    FragU f;
    f.q[0] = *(const u32x4*)(w + n * stride + kh);
    f.q[1] = *(const u32x4*)(w + n * stride + kh + 8);
    return f.v;
}

// Same B fragment but streamed from a global fp32 row-major weight (L2-resident),
// packed to bf16 on the fly.
__device__ __forceinline__ v16bf load_bfrag_gf32(const float* __restrict__ w, int stride,
                                                 int n0, int k0, int lane) {
    const float4* p = (const float4*)(w + (n0 + (lane & 15)) * stride
                                        + k0 + ((lane >> 4) << 4));
    float4 a = p[0], b = p[1], c = p[2], d = p[3];
    FragU f;
    f.q[0][0] = pk2(a.x, a.y); f.q[0][1] = pk2(a.z, a.w);
    f.q[0][2] = pk2(b.x, b.y); f.q[0][3] = pk2(b.z, b.w);
    f.q[1][0] = pk2(c.x, c.y); f.q[1][1] = pk2(c.z, c.w);
    f.q[1][2] = pk2(d.x, d.y); f.q[1][3] = pk2(d.z, d.w);
    return f.v;
}

#define WMMA_BF16(A, B, C) \
    __builtin_amdgcn_wmma_f32_16x16x32_bf16(false, (A), false, (B), (short)0, (C), false, false)

// ================================================================
// Fully fused: LSTM0 -> LSTM1 -> Linear+BN -> Linear+ReLU -> Linear
// One workgroup = 16 batch rows. 8 waves = (gate 0..3) x (col-half 0..1).
// ================================================================
__global__ __launch_bounds__(256)
void tb_lstm_fused(const float* __restrict__ x,
                   const float* __restrict__ Wih0, const float* __restrict__ Whh0,
                   const float* __restrict__ bih0, const float* __restrict__ bhh0,
                   const float* __restrict__ Wih1, const float* __restrict__ Whh1,
                   const float* __restrict__ bih1, const float* __restrict__ bhh1,
                   const float* __restrict__ Wp,   const float* __restrict__ bp,
                   const float* __restrict__ bng,  const float* __restrict__ bnb,
                   const float* __restrict__ bnm,  const float* __restrict__ bnv,
                   const float* __restrict__ W1,   const float* __restrict__ b1,
                   const float* __restrict__ W2,   const float* __restrict__ b2,
                   float* __restrict__ out)
{
    extern __shared__ char smem[];
    unsigned short* wbuf  = (unsigned short*)(smem + OFF_WBUF);   // W_hh (bf16)
    unsigned short* wih0  = (unsigned short*)(smem + OFF_WIH0);   // W_ih0 padded K=32
    unsigned short* h1seq = (unsigned short*)(smem + OFF_H1SEQ);  // layer-0 outputs
    float*          gact  = (float*)(smem + OFF_GACT);            // activated gates
    float*          cbuf  = (float*)(smem + OFF_C);
    unsigned short* hbuf  = (unsigned short*)(smem + OFF_H);
    unsigned short* xbuf  = (unsigned short*)(smem + OFF_XBUF);
    float*          bias0 = (float*)(smem + OFF_B0);
    float*          bias1 = (float*)(smem + OFF_B1);

    const int tid  = threadIdx.x;
    const int lane = tid & 31;
    const int wave = tid >> 5;
    const int gate = wave >> 1;   // 0:i 1:f 2:g 3:o
    const int half = wave & 1;
    const int row0 = blockIdx.x * BM;

    // Branch-free gate nonlinearity:
    //   sigmoid: y =     rcp(1 + exp(-x))
    //   tanh:    y = 2 * rcp(1 + exp(-2x)) - 1
    // selected per wave by three cndmask constants (no EXEC-mask branches).
    const float g_in  = (gate == 2) ? -2.0f : -1.0f;   // exp input scale
    const float g_mul = (gate == 2) ?  2.0f :  1.0f;   // output scale
    const float g_add = (gate == 2) ? -1.0f :  0.0f;   // output offset

    // ---- stage W_hh0 (fp32 -> bf16), padded W_ih0, combined biases, zero state ----
    for (int i4 = tid; i4 < (G4 * H1) / 4; i4 += 256) {
        float4 w = ((const float4*)Whh0)[i4];
        ((unsigned int*)(void*)wbuf)[i4 * 2 + 0] = pk2(w.x, w.y);
        ((unsigned int*)(void*)wbuf)[i4 * 2 + 1] = pk2(w.z, w.w);
    }
    for (int i = tid; i < G4 * 32; i += 256) {
        int g = i >> 5, k = i & 31;
        wih0[i] = (k < NFEAT) ? f2bf(Wih0[g * NFEAT + k]) : (unsigned short)0;
    }
    for (int i = tid; i < G4; i += 256) {
        bias0[i] = bih0[i] + bhh0[i];
        bias1[i] = bih1[i] + bhh1[i];
    }
    for (int i = tid; i < BM * H1; i += 256) { cbuf[i] = 0.0f; hbuf[i] = 0; }
    __syncthreads();

    // ======================= Phase A: LSTM layer 0 =======================
    for (int t = 0; t < SEQ; ++t) {
        // stage x_t tile as bf16, K padded 25 -> 32
        for (int i = tid; i < BM * 32; i += 256) {
            int m = i >> 5, k = i & 31;
            xbuf[i] = (k < NFEAT) ? f2bf(x[((row0 + m) * SEQ + t) * NFEAT + k])
                                  : (unsigned short)0;
        }
        __syncthreads();

        v16bf ax = load_afrag(xbuf, 32, 0, lane);
        v16bf ah[4];
#pragma unroll
        for (int kc = 0; kc < 4; ++kc) ah[kc] = load_afrag(hbuf, H1, kc * 32, lane);

#pragma unroll
        for (int i = 0; i < 4; ++i) {
            const int n0 = gate * 128 + half * 64 + i * 16;
            // two independent WMMA chains (interleaved K chunks) -> more ILP
            v8f accx = {0.f, 0.f, 0.f, 0.f, 0.f, 0.f, 0.f, 0.f};
            v8f acch = {0.f, 0.f, 0.f, 0.f, 0.f, 0.f, 0.f, 0.f};
            accx = WMMA_BF16(ax,    load_bfrag(wih0, 32, n0, 0,  lane), accx);
            acch = WMMA_BF16(ah[0], load_bfrag(wbuf, H1, n0, 0,  lane), acch);
            accx = WMMA_BF16(ah[1], load_bfrag(wbuf, H1, n0, 32, lane), accx);
            acch = WMMA_BF16(ah[2], load_bfrag(wbuf, H1, n0, 64, lane), acch);
            accx = WMMA_BF16(ah[3], load_bfrag(wbuf, H1, n0, 96, lane), accx);

            // bias + gate nonlinearity in-register, scatter to LDS
            const int   n  = n0 + (lane & 15);
            const float bb = bias0[n];
            const int   mb = (lane >> 4) << 3;
#pragma unroll
            for (int r = 0; r < 8; ++r) {
                float v = accx[r] + acch[r] + bb;
                v = g_mul * fast_rcp(1.0f + __expf(g_in * v)) + g_add;
                gact[(mb + r) * G4 + n] = v;
            }
        }
        __syncthreads();

        // cell / hidden update (i,f,g,o PyTorch order)
        for (int i = tid; i < BM * H1; i += 256) {
            int m = i >> 7, j = i & 127;
            float ig = gact[m * G4 + j];
            float fg = gact[m * G4 + 128 + j];
            float gg = gact[m * G4 + 256 + j];
            float og = gact[m * G4 + 384 + j];
            float c  = fg * cbuf[i] + ig * gg;
            cbuf[i]  = c;
            unsigned short hb = f2bf(og * tanh_fast(c));
            hbuf[i] = hb;
            h1seq[t * (BM * H1) + i] = hb;
        }
        __syncthreads();
    }

    // ---- restage wbuf with W_hh1; reset recurrent state ----
    for (int i4 = tid; i4 < (G4 * H1) / 4; i4 += 256) {
        float4 w = ((const float4*)Whh1)[i4];
        ((unsigned int*)(void*)wbuf)[i4 * 2 + 0] = pk2(w.x, w.y);
        ((unsigned int*)(void*)wbuf)[i4 * 2 + 1] = pk2(w.z, w.w);
    }
    for (int i = tid; i < BM * H1; i += 256) { cbuf[i] = 0.0f; hbuf[i] = 0; }
    __syncthreads();

    // ======================= Phase B: LSTM layer 1 =======================
    for (int t = 0; t < SEQ; ++t) {
        const unsigned short* hx = h1seq + t * (BM * H1);
        v16bf as[4], ah[4];
#pragma unroll
        for (int kc = 0; kc < 4; ++kc) {
            as[kc] = load_afrag(hx,   H1, kc * 32, lane);
            ah[kc] = load_afrag(hbuf, H1, kc * 32, lane);
        }

#pragma unroll
        for (int i = 0; i < 4; ++i) {
            const int n0 = gate * 128 + half * 64 + i * 16;
            v8f acca = {0.f, 0.f, 0.f, 0.f, 0.f, 0.f, 0.f, 0.f};
            v8f accb = {0.f, 0.f, 0.f, 0.f, 0.f, 0.f, 0.f, 0.f};
#pragma unroll
            for (int kc = 0; kc < 4; ++kc)   // h1seq @ Wih1^T  (B streamed via L2)
                acca = WMMA_BF16(as[kc], load_bfrag_gf32(Wih1, H1, n0, kc * 32, lane), acca);
#pragma unroll
            for (int kc = 0; kc < 4; ++kc)   // h @ Whh1^T      (B from LDS)
                accb = WMMA_BF16(ah[kc], load_bfrag(wbuf, H1, n0, kc * 32, lane), accb);

            const int   n  = n0 + (lane & 15);
            const float bb = bias1[n];
            const int   mb = (lane >> 4) << 3;
#pragma unroll
            for (int r = 0; r < 8; ++r) {
                float v = acca[r] + accb[r] + bb;
                v = g_mul * fast_rcp(1.0f + __expf(g_in * v)) + g_add;
                gact[(mb + r) * G4 + n] = v;
            }
        }
        __syncthreads();

        for (int i = tid; i < BM * H1; i += 256) {
            int m = i >> 7, j = i & 127;
            float ig = gact[m * G4 + j];
            float fg = gact[m * G4 + 128 + j];
            float gg = gact[m * G4 + 256 + j];
            float og = gact[m * G4 + 384 + j];
            float c  = fg * cbuf[i] + ig * gg;
            cbuf[i]  = c;
            hbuf[i]  = f2bf(og * tanh_fast(c));
        }
        __syncthreads();
    }

    // ======================= Phase C: head (fp32 scalar) =======================
    float* ctx = gact;             // [16][64]   (reuse gate buffer)
    float* hid = gact + BM * H2;   // [16][32]

    for (int i = tid; i < BM * H2; i += 256) {       // hidden_proj + BatchNorm
        int m = i >> 6, n = i & 63;
        float s = bp[n];
        for (int j = 0; j < H1; ++j) s += bf2f(hbuf[m * H1 + j]) * Wp[n * H1 + j];
        float sc = bng[n] * __builtin_amdgcn_rsqf(bnv[n] + 1e-5f);
        ctx[i] = (s - bnm[n]) * sc + bnb[n];
    }
    __syncthreads();
    for (int i = tid; i < BM * DH; i += 256) {       // Linear(64->32) + ReLU
        int m = i >> 5, n = i & 31;
        float s = b1[n];
        for (int j = 0; j < H2; ++j) s += ctx[m * H2 + j] * W1[n * H2 + j];
        hid[i] = fmaxf(s, 0.0f);
    }
    __syncthreads();
    for (int i = tid; i < BM * NC; i += 256) {       // Linear(32->3)
        int m = i / NC, n = i - m * NC;
        float s = b2[n];
        for (int j = 0; j < DH; ++j) s += hid[m * DH + j] * W2[n * DH + j];
        out[(row0 + m) * NC + n] = s;
    }
}

extern "C" void kernel_launch(void* const* d_in, const int* in_sizes, int n_in,
                              void* d_out, int out_size, void* d_ws, size_t ws_size,
                              hipStream_t stream) {
    (void)in_sizes; (void)n_in; (void)out_size; (void)d_ws; (void)ws_size;
    const float* x    = (const float*)d_in[0];
    const float* Wih0 = (const float*)d_in[1];
    const float* Whh0 = (const float*)d_in[2];
    const float* bih0 = (const float*)d_in[3];
    const float* bhh0 = (const float*)d_in[4];
    const float* Wih1 = (const float*)d_in[5];
    const float* Whh1 = (const float*)d_in[6];
    const float* bih1 = (const float*)d_in[7];
    const float* bhh1 = (const float*)d_in[8];
    const float* Wp   = (const float*)d_in[9];
    const float* bp   = (const float*)d_in[10];
    const float* bng  = (const float*)d_in[11];
    const float* bnb  = (const float*)d_in[12];
    const float* bnm  = (const float*)d_in[13];
    const float* bnv  = (const float*)d_in[14];
    const float* W1   = (const float*)d_in[15];
    const float* b1   = (const float*)d_in[16];
    const float* W2   = (const float*)d_in[17];
    const float* b2   = (const float*)d_in[18];

    dim3 grid(BATCH / BM), block(256);
    tb_lstm_fused<<<grid, block, LDS_BYTES, stream>>>(
        x, Wih0, Whh0, bih0, bhh0, Wih1, Whh1, bih1, bhh1,
        Wp, bp, bng, bnb, bnm, bnv, W1, b1, W2, b2, (float*)d_out);
}
